// EnhancedCrossAttention3D_18476949307799
// MI455X (gfx1250) — matrix operand order repaired and checked by reading, not compiled, not run on hardware
//
#include <hip/hip_runtime.h>
#include <hip/hip_bf16.h>

// Flash-attention for EnhancedCrossAttention3D on gfx1250 (MI455X).
// B=2, C=64, N=8192. f16 WMMA (v_wmma_f32_16x16x32_f16), f32 accumulate.
// Row-max via v_permlane16_b32 (VALU), row-sum via WMMA against ones-matrix.

typedef __attribute__((ext_vector_type(16))) _Float16 v16h;
typedef __attribute__((ext_vector_type(8)))  _Float16 v8h;
typedef __attribute__((ext_vector_type(8)))  float    v8f;

#define NTOK 8192
#define CCH  64
#define NB   2
#define SM_SCALE 0.125f   // 1/sqrt(64)

#define DS_FENCE() asm volatile("s_wait_dscnt 0" ::: "memory")

static __device__ inline v16h cat8(v8h lo, v8h hi) {
  v16h r;
#pragma unroll
  for (int i = 0; i < 8; ++i) { r[i] = lo[i]; r[8 + i] = hi[i]; }
  return r;
}

static __device__ inline v8f wmma16(v16h a, v16h b, v8f c) {
  // D = A(16x32 f16) * B(32x16 f16) + C(16x16 f32)
  return __builtin_amdgcn_wmma_f32_16x16x32_f16(false, a, false, b, (short)0, c, false, false);
}

// max-reduce across the 16 lanes of each half-wave row group using
// v_permlane16_b32 with XOR lane selectors (pure VALU, no LDS traffic).
static __device__ inline float rowmax16(float v) {
  int x = __float_as_int(v), y;
  y = __builtin_amdgcn_permlane16(x, x, (int)0x67452301, (int)0xEFCDAB89, false, false); // xor 1
  x = __float_as_int(fmaxf(__int_as_float(x), __int_as_float(y)));
  y = __builtin_amdgcn_permlane16(x, x, (int)0x54761032, (int)0xDCFE98BA, false, false); // xor 2
  x = __float_as_int(fmaxf(__int_as_float(x), __int_as_float(y)));
  y = __builtin_amdgcn_permlane16(x, x, (int)0x32107654, (int)0xBA98FEDC, false, false); // xor 4
  x = __float_as_int(fmaxf(__int_as_float(x), __int_as_float(y)));
  y = __builtin_amdgcn_permlane16(x, x, (int)0xFEDCBA98, (int)0x76543210, false, false); // xor 8
  x = __float_as_int(fmaxf(__int_as_float(x), __int_as_float(y)));
  return __int_as_float(x);
}

// ---------------------------------------------------------------------------
// QKV projection: q[b,n,c] (f16), k[b,n,c] (f16), v_t[b,c,n] (f16)
// ---------------------------------------------------------------------------
__global__ __launch_bounds__(256) void proj_qkv_kernel(
    const float* __restrict__ x1, const float* __restrict__ x2,
    const float* __restrict__ Wq, const float* __restrict__ bq,
    const float* __restrict__ Wk, const float* __restrict__ bk,
    const float* __restrict__ Wv, const float* __restrict__ bv,
    _Float16* __restrict__ qh, _Float16* __restrict__ kh, _Float16* __restrict__ vt)
{
  const int b  = blockIdx.z;
  const int co = blockIdx.y;                       // output channel
  const int n  = blockIdx.x * 256 + threadIdx.x;   // voxel
  const float* x1b = x1 + (size_t)b * CCH * NTOK;
  const float* x2b = x2 + (size_t)b * CCH * NTOK;
  float aq = 0.f, ak = 0.f, av = 0.f;
#pragma unroll 4
  for (int c = 0; c < CCH; ++c) {
    const float wq = Wq[co * CCH + c];
    const float wk = Wk[co * CCH + c];
    const float wv = Wv[co * CCH + c];
    const float xa = x1b[(size_t)c * NTOK + n];
    const float xb = x2b[(size_t)c * NTOK + n];
    aq = fmaf(wq, xa, aq);
    ak = fmaf(wk, xb, ak);
    av = fmaf(wv, xb, av);
  }
  qh[((size_t)b * NTOK + n) * CCH + co] = (_Float16)(aq + bq[co]);
  kh[((size_t)b * NTOK + n) * CCH + co] = (_Float16)(ak + bk[co]);
  vt[((size_t)b * CCH + co) * NTOK + n] = (_Float16)(av + bv[co]);
}

__global__ __launch_bounds__(256) void cvt_wp_kernel(const float* __restrict__ Wp,
                                                     _Float16* __restrict__ wph)
{
  const int i = blockIdx.x * 256 + threadIdx.x;
  if (i < CCH * CCH) wph[i] = (_Float16)Wp[i];
}

// ---------------------------------------------------------------------------
// Flash attention + fused output projection.
// 1 wave owns 16 query rows; key tile = 64; 4 waves per block.
// ---------------------------------------------------------------------------
__global__ __launch_bounds__(128) void fa_kernel(
    const _Float16* __restrict__ qh, const _Float16* __restrict__ kh,
    const _Float16* __restrict__ vt, const _Float16* __restrict__ wph,
    const float* __restrict__ bp, float* __restrict__ out)
{
  __shared__ _Float16 lds[4][16 * 64];   // per-wave scratch (P: 16x64, O: 16x64)

  const int wave = threadIdx.x >> 5;
  const int lane = threadIdx.x & 31;
  const int hf   = lane >> 4;            // half-wave: 0 or 1
  const int l16  = lane & 15;

  const int tile  = blockIdx.x * 4 + wave;       // 0 .. B*(N/16)-1
  const int b     = tile / (NTOK / 16);
  const int qbase = (tile % (NTOK / 16)) * 16;

  _Float16* Pw = &lds[wave][0];

  // ---- ones B-matrix fragment (32x16 of 1.0) for WMMA row-sums ------------
  v16h vones;
#pragma unroll
  for (int i = 0; i < 16; ++i) vones[i] = (_Float16)1.0f;

  // ---- load A-fragments of Q (16 rows x 64 ch = two K=32 steps) -----------
  const _Float16* qrow = qh + ((size_t)b * NTOK + qbase + l16) * CCH;
  v16h aq[2];
#pragma unroll
  for (int s = 0; s < 2; ++s) {
    const _Float16* p = qrow + 32 * s + 16 * hf;
    aq[s] = *(const v16h*)p;
  }

  // ---- running stats + O accumulators (16 rows x 64 ch, 4 col-blocks) -----
  float m[8], l[8];
  v8f o[4];
#pragma unroll
  for (int r = 0; r < 8; ++r) { m[r] = -1e30f; l[r] = 0.f; }
#pragma unroll
  for (int cb = 0; cb < 4; ++cb) o[cb] = (v8f){};

  const _Float16* kb_base = kh + (size_t)b * NTOK * CCH;
  const _Float16* vb_base = vt + (size_t)b * CCH * NTOK;

#pragma unroll 1
  for (int kt = 0; kt < NTOK; kt += 64) {
    const _Float16* kb = kb_base + (size_t)kt * CCH;

    // prefetch next key/value tiles (L2-resident stream)
    if (kt + 64 < NTOK) {
      __builtin_prefetch(kb + (size_t)64 * CCH + (size_t)lane * 2 * CCH, 0, 1);
      __builtin_prefetch(vb_base + kt + 64 + (size_t)lane * 2 * NTOK, 0, 1);
    }

    // ---- S = Q * K^T : 4 col-tiles of 16 keys, K-dim = 64 -----------------
    v8f s[4];
#pragma unroll
    for (int j = 0; j < 4; ++j) {
      const _Float16* kc = kb + (size_t)(j * 16 + l16) * CCH;
      v16h b0 = *(const v16h*)(kc + 16 * hf);
      v16h b1 = *(const v16h*)(kc + 32 + 16 * hf);
      v8f acc = (v8f){};
      acc = wmma16(aq[0], b0, acc);
      acc = wmma16(aq[1], b1, acc);
      s[j] = acc;
    }

    // ---- online softmax: max via permlane, P staged to LDS ----------------
    float alpha[8];
#pragma unroll
    for (int r = 0; r < 8; ++r) {
      const float x0 = s[0][r] * SM_SCALE;
      const float x1 = s[1][r] * SM_SCALE;
      const float x2 = s[2][r] * SM_SCALE;
      const float x3 = s[3][r] * SM_SCALE;
      float t = fmaxf(fmaxf(x0, x1), fmaxf(x2, x3));
      t = rowmax16(t);
      const float mn = fmaxf(m[r], t);
      alpha[r] = __expf(m[r] - mn);
      m[r] = mn;
      const float p0 = __expf(x0 - mn);
      const float p1 = __expf(x1 - mn);
      const float p2 = __expf(x2 - mn);
      const float p3 = __expf(x3 - mn);
      const int row = (r + 8 * hf) * 64;
      Pw[row + l16]      = (_Float16)p0;
      Pw[row + 16 + l16] = (_Float16)p1;
      Pw[row + 32 + l16] = (_Float16)p2;
      Pw[row + 48 + l16] = (_Float16)p3;
    }

    // ---- rescale O by alpha (same C/D layout => per-lane row match) -------
#pragma unroll
    for (int cb = 0; cb < 4; ++cb)
#pragma unroll
      for (int r = 0; r < 8; ++r) o[cb][r] *= alpha[r];

    DS_FENCE();  // same-wave DS in-order; wait for cross-lane P data

    // ---- reload P as two A fragments (16x32 each) -------------------------
    v16h ap[2];
#pragma unroll
    for (int s2 = 0; s2 < 2; ++s2) {
      const _Float16* pp = &Pw[l16 * 64 + 32 * s2 + 16 * hf];
      ap[s2] = *(const v16h*)pp;
    }

    // ---- row-sum of P via WMMA against ones (replaces shuffle reduce) -----
    v8f srow = (v8f){};
    srow = wmma16(ap[0], vones, srow);
    srow = wmma16(ap[1], vones, srow);
#pragma unroll
    for (int r = 0; r < 8; ++r) l[r] = l[r] * alpha[r] + srow[r];

    // ---- O += P * V : 4 col-blocks of 16 channels, K-dim = 64 keys --------
    const _Float16* vb = vb_base + kt + 16 * hf;
#pragma unroll
    for (int cb = 0; cb < 4; ++cb) {
      const _Float16* vc = vb + (size_t)(cb * 16 + l16) * NTOK;
      o[cb] = wmma16(ap[0], *(const v16h*)(vc), o[cb]);
      o[cb] = wmma16(ap[1], *(const v16h*)(vc + 32), o[cb]);
    }
  }

  // ---- normalize, stage O (16x64 f16) for the output projection -----------
#pragma unroll
  for (int r = 0; r < 8; ++r) {
    const float linv = 1.0f / l[r];
#pragma unroll
    for (int cb = 0; cb < 4; ++cb)
      Pw[(r + 8 * hf) * 64 + cb * 16 + l16] = (_Float16)(o[cb][r] * linv);
  }
  DS_FENCE();

  v16h ao[2];
#pragma unroll
  for (int s = 0; s < 2; ++s) {
    const _Float16* p = &Pw[l16 * 64 + 32 * s + 16 * hf];
    ao[s] = *(const v16h*)p;
  }

  // ---- out[b,co,n] = Wp * O^T + bp  (4 col-blocks of co) ------------------
#pragma unroll
  for (int cb = 0; cb < 4; ++cb) {
    v8f d = (v8f){};
#pragma unroll
    for (int s = 0; s < 2; ++s) {
      v16h bw = *(const v16h*)(wph + (size_t)(cb * 16 + l16) * CCH + 32 * s + 16 * hf);
      d = wmma16(ao[s], bw, d);
    }
    const int co = cb * 16 + l16;
    const float bias = bp[co];
    float* ob = out + ((size_t)b * CCH + co) * NTOK + qbase + 8 * hf;
#pragma unroll
    for (int r = 0; r < 8; ++r) ob[r] = d[r] + bias;
  }
}

// ---------------------------------------------------------------------------
extern "C" void kernel_launch(void* const* d_in, const int* in_sizes, int n_in,
                              void* d_out, int out_size, void* d_ws, size_t ws_size,
                              hipStream_t stream) {
  const float* x1 = (const float*)d_in[0];
  const float* x2 = (const float*)d_in[1];
  const float* Wq = (const float*)d_in[2];
  const float* bq = (const float*)d_in[3];
  const float* Wk = (const float*)d_in[4];
  const float* bk = (const float*)d_in[5];
  const float* Wv = (const float*)d_in[6];
  const float* bv = (const float*)d_in[7];
  const float* Wp = (const float*)d_in[8];
  const float* bp = (const float*)d_in[9];
  float* out = (float*)d_out;

  _Float16* qh  = (_Float16*)d_ws;                       // 2*8192*64 halves
  _Float16* kh  = qh + (size_t)NB * NTOK * CCH;
  _Float16* vt  = kh + (size_t)NB * NTOK * CCH;
  _Float16* wph = vt + (size_t)NB * NTOK * CCH;          // 64*64 halves

  proj_qkv_kernel<<<dim3(NTOK / 256, CCH, NB), 256, 0, stream>>>(
      x1, x2, Wq, bq, Wk, bk, Wv, bv, qh, kh, vt);
  cvt_wp_kernel<<<dim3((CCH * CCH + 255) / 256), 256, 0, stream>>>(Wp, wph);

  const int ntiles = NB * (NTOK / 16);   // 1024 q-row tiles
  fa_kernel<<<dim3(ntiles / 4), 128, 0, stream>>>(qh, kh, vt, wph, bp, out);
}
